// KohaBlock_87849261072486
// MI455X (gfx1250) — compile-verified, matrix operand order checked by reading, not run on hardware
//
#include <hip/hip_runtime.h>

#define EMB 1024
#define NHEAD 16
#define HS 64
#define RF 64
#define BTOT 2048

typedef unsigned int u32;
typedef unsigned short u16;
typedef __attribute__((ext_vector_type(4)))  u32   u32x4;
typedef __attribute__((ext_vector_type(8)))  u32   u32x8;
typedef __attribute__((ext_vector_type(16))) __bf16 v16bf;
typedef __attribute__((ext_vector_type(8)))  float v8f;

// ---------- helpers ----------
__device__ __forceinline__ u16 f2bf(float f) {
  u32 u = __builtin_bit_cast(u32, f);
  u32 r = u + 0x7FFFu + ((u >> 16) & 1u);   // round-to-nearest-even
  return (u16)(r >> 16);
}

__device__ __forceinline__ v8f vzero8() { v8f v{}; return v; }

// Load one 16x32 bf16 WMMA fragment (A: lane=M-row, B: lane=N-col) from a
// row-major [rows][ld] bf16 matrix. Per ISA 7.12.2: lanes 0-15 take K=k0+0..7
// and k0+16..23; lanes 16-31 take K=k0+8..15 and k0+24..31.
__device__ __forceinline__ v16bf load_frag(const u16* base, int row0, int ld,
                                           int k0, int lane) {
  const int rsel = lane & 15;
  const int hi   = (lane >> 4) & 1;
  const u16* p = base + (size_t)(row0 + rsel) * ld + k0 + hi * 8;
  const u32x4 a = *(const u32x4*)(p);
  const u32x4 b = *(const u32x4*)(p + 16);
  u32x8 c;
  c[0]=a[0]; c[1]=a[1]; c[2]=a[2]; c[3]=a[3];
  c[4]=b[0]; c[5]=b[1]; c[6]=b[2]; c[7]=b[3];
  return __builtin_bit_cast(v16bf, c);
}

#define WMMA_BF16(A,B,C) __builtin_amdgcn_wmma_f32_16x16x32_bf16(false,(A),false,(B),(short)0,(C),false,false)

// ---------- prep kernels ----------
__global__ void cvt_bf16_v4(const float4* __restrict__ in, ushort4* __restrict__ out, long n4) {
  long i = (long)blockIdx.x * blockDim.x + threadIdx.x;
  long stride = (long)gridDim.x * blockDim.x;
  for (; i < n4; i += stride) {
    float4 v = in[i];
    ushort4 o;
    o.x = f2bf(v.x); o.y = f2bf(v.y); o.z = f2bf(v.z); o.w = f2bf(v.w);
    out[i] = o;
  }
}

// in: [batch][64][64] f32 -> out[batch][j][i] bf16 (transpose inner 64x64)
__global__ void transpose64_bf16(const float* __restrict__ in, u16* __restrict__ out) {
  const long b = blockIdx.x;
  for (int e = threadIdx.x; e < 4096; e += blockDim.x) {
    int i = e >> 6, j = e & 63;
    out[b * 4096 + (long)j * 64 + i] = f2bf(in[b * 4096 + (long)i * 64 + j]);
  }
}

// ---------- kernel 1: q path ----------
__global__ __launch_bounds__(256) void koha_query(
    const float* __restrict__ x, const u16* __restrict__ qk,
    const u16* __restrict__ qvt, float* __restrict__ qp, float* __restrict__ qn) {
  extern __shared__ char lds_raw[];
  u16* xt = (u16*)lds_raw;                          // [16][1032] bf16
  const int tid  = threadIdx.x;
  const int lane = tid & 31;
  const int wave = tid >> 5;
  u16* pmp = (u16*)(lds_raw + 16 * 1032 * 2) + wave * 2 * (16 * 72);
  u16* pmn = pmp + 16 * 72;

  const int b0 = blockIdx.x * 16;
  for (int idx = tid; idx < 16 * 256; idx += 256) {
    int row = idx >> 8, g = idx & 255;
    float4 v = *(const float4*)(x + (size_t)(b0 + row) * EMB + g * 4);
    u16* d = xt + row * 1032 + g * 4;
    d[0]=f2bf(v.x); d[1]=f2bf(v.y); d[2]=f2bf(v.z); d[3]=f2bf(v.w);
  }
  __syncthreads();

  const int n15 = lane & 15;
  const int half = (lane >> 4) & 1;
  for (int hh = 0; hh < 2; ++hh) {
    const int h = wave * 2 + hh;
    v8f acc[4];
    #pragma unroll
    for (int nt = 0; nt < 4; ++nt) acc[nt] = vzero8();
    const u16* qkb = qk + (size_t)h * HS * EMB;
    __builtin_prefetch(qkb + (size_t)lane * 128, 0, 0);
    for (int k0 = 0; k0 < EMB; k0 += 32) {
      v16bf Af = load_frag(xt, 0, 1032, k0, lane);
      #pragma unroll
      for (int nt = 0; nt < 4; ++nt) {
        v16bf Bf = load_frag(qkb, nt * 16, EMB, k0, lane);
        acc[nt] = WMMA_BF16(Af, Bf, acc[nt]);
      }
    }
    // +/- softmax over n per row (row = half*8 + i; 16 lanes hold a row)
    #pragma unroll
    for (int i = 0; i < 8; ++i) {
      int row = half * 8 + i;
      float mx = acc[0][i], mn = acc[0][i];
      #pragma unroll
      for (int nt = 1; nt < 4; ++nt) { mx = fmaxf(mx, acc[nt][i]); mn = fminf(mn, acc[nt][i]); }
      #pragma unroll
      for (int off = 1; off < 16; off <<= 1) {
        mx = fmaxf(mx, __shfl_xor(mx, off, 32));
        mn = fminf(mn, __shfl_xor(mn, off, 32));
      }
      float ep[4], en[4], sp_ = 0.f, sn_ = 0.f;
      #pragma unroll
      for (int nt = 0; nt < 4; ++nt) {
        ep[nt] = __expf(acc[nt][i] - mx);  sp_ += ep[nt];
        en[nt] = __expf(-acc[nt][i] + mn); sn_ += en[nt];
      }
      #pragma unroll
      for (int off = 1; off < 16; off <<= 1) {
        sp_ += __shfl_xor(sp_, off, 32);
        sn_ += __shfl_xor(sn_, off, 32);
      }
      float ip = 1.f / sp_, in_ = 1.f / sn_;
      #pragma unroll
      for (int nt = 0; nt < 4; ++nt) {
        pmp[row * 72 + nt * 16 + n15] = f2bf(ep[nt] * ip);
        pmn[row * 72 + nt * 16 + n15] = f2bf(en[nt] * in_);
      }
    }
    __builtin_amdgcn_wave_barrier();
    // GEMM2: (16x64) @ qv_T (N-major 64x64)
    const u16* qvb = qvt + (size_t)h * HS * HS;
    v8f ap[4], an[4];
    #pragma unroll
    for (int nt = 0; nt < 4; ++nt) { ap[nt] = vzero8(); an[nt] = vzero8(); }
    #pragma unroll
    for (int k0 = 0; k0 < 64; k0 += 32) {
      v16bf Bf[4];
      #pragma unroll
      for (int nt = 0; nt < 4; ++nt) Bf[nt] = load_frag(qvb, nt * 16, 64, k0, lane);
      v16bf Ap = load_frag(pmp, 0, 72, k0, lane);
      v16bf An = load_frag(pmn, 0, 72, k0, lane);
      #pragma unroll
      for (int nt = 0; nt < 4; ++nt) { ap[nt] = WMMA_BF16(Ap, Bf[nt], ap[nt]); an[nt] = WMMA_BF16(An, Bf[nt], an[nt]); }
    }
    #pragma unroll
    for (int nt = 0; nt < 4; ++nt)
      #pragma unroll
      for (int i = 0; i < 8; ++i) {
        size_t o = ((size_t)(b0 + half * 8 + i) * NHEAD + h) * HS + nt * 16 + n15;
        qp[o] = ap[nt][i];
        qn[o] = an[nt][i];
      }
    __builtin_amdgcn_wave_barrier();
  }
}

// ---------- kernel 2: main (k path + scores) ----------
// Per-wave LDS scratch layout: sc = [64][68] f32 score tile; the normalized
// bf16 tile (ksm, ld=136 u16) is overlaid row-locally on the same storage.
__global__ __launch_bounds__(256) void koha_scores(
    const float* __restrict__ z, const u16* __restrict__ kk,
    const u16* __restrict__ kvt, const float* __restrict__ qp,
    const float* __restrict__ qn, float* __restrict__ sp, float* __restrict__ sn) {
  extern __shared__ char lds_raw[];
  u16* zt = (u16*)lds_raw;                                   // [64][1032]
  const int tid  = threadIdx.x;
  const int lane = tid & 31;
  const int wave = tid >> 5;
  float* sc = (float*)(lds_raw + 64 * 1032 * 2) + wave * (64 * 68);
  u16* ksm = (u16*)sc;                                       // overlay, ld=136

  const int r  = blockIdx.y;
  const int b0 = blockIdx.x * 64;

  for (int idx = tid; idx < 64 * 256; idx += 256) {
    int row = idx >> 8, g = idx & 255;
    float4 v = *(const float4*)(z + (((size_t)(b0 + row)) * RF + r) * EMB + g * 4);
    u16* d = zt + row * 1032 + g * 4;
    d[0]=f2bf(v.x); d[1]=f2bf(v.y); d[2]=f2bf(v.z); d[3]=f2bf(v.w);
  }
  __syncthreads();

  const int n15 = lane & 15;
  const int half = (lane >> 4) & 1;
  for (int hh = 0; hh < 2; ++hh) {
    const int h = wave * 2 + hh;
    const u16* kkb = kk + (((size_t)h * RF + r) * HS) * EMB;
    __builtin_prefetch(kkb + (size_t)lane * 256, 0, 0);
    // GEMM1, nt-outer pass: key_keys stream read exactly once from L2.
    #pragma unroll
    for (int nt = 0; nt < 4; ++nt) {
      v8f acc[4];
      #pragma unroll
      for (int mt = 0; mt < 4; ++mt) acc[mt] = vzero8();
      for (int k0 = 0; k0 < EMB; k0 += 32) {
        v16bf Bf = load_frag(kkb, nt * 16, EMB, k0, lane);
        #pragma unroll
        for (int mt = 0; mt < 4; ++mt) {
          v16bf Af = load_frag(zt, mt * 16, 1032, k0, lane);
          acc[mt] = WMMA_BF16(Af, Bf, acc[mt]);
        }
      }
      #pragma unroll
      for (int mt = 0; mt < 4; ++mt)
        #pragma unroll
        for (int i = 0; i < 8; ++i)
          sc[(mt * 16 + half * 8 + i) * 68 + nt * 16 + n15] = acc[mt][i];
    }
    __builtin_amdgcn_wave_barrier();
    // per-lane row softmax over n; write bf16 A-tile into row-local overlay
    #pragma unroll
    for (int rr = 0; rr < 2; ++rr) {
      int row = lane * 2 + rr;
      const float* sr = sc + row * 68;
      float v[HS];
      #pragma unroll
      for (int g = 0; g < 16; ++g) {
        float4 t = *(const float4*)(sr + g * 4);
        v[g*4+0]=t.x; v[g*4+1]=t.y; v[g*4+2]=t.z; v[g*4+3]=t.w;
      }
      float mx = v[0];
      #pragma unroll
      for (int j = 1; j < HS; ++j) mx = fmaxf(mx, v[j]);
      float s = 0.f;
      #pragma unroll
      for (int j = 0; j < HS; ++j) { v[j] = __expf(v[j] - mx); s += v[j]; }
      float inv = 1.f / s;
      u16* kr = ksm + row * 136;
      #pragma unroll
      for (int g = 0; g < 16; ++g) {
        ushort4 o;
        o.x = f2bf(v[g*4+0]*inv); o.y = f2bf(v[g*4+1]*inv);
        o.z = f2bf(v[g*4+2]*inv); o.w = f2bf(v[g*4+3]*inv);
        *(ushort4*)(kr + g * 4) = o;
      }
    }
    __builtin_amdgcn_wave_barrier();
    // GEMM2 + q.k dot, mt-outer (kvt tile is 8KB, L2-hot)
    const u16* kvb = kvt + (((size_t)h * RF + r) * HS) * HS;
    const float scale = 0.125f;
    #pragma unroll
    for (int mt = 0; mt < 4; ++mt) {
      v8f kp[4];
      #pragma unroll
      for (int nt = 0; nt < 4; ++nt) kp[nt] = vzero8();
      #pragma unroll
      for (int k0 = 0; k0 < 64; k0 += 32) {
        v16bf Af = load_frag(ksm, mt * 16, 136, k0, lane);
        #pragma unroll
        for (int nt = 0; nt < 4; ++nt) {
          v16bf Bf = load_frag(kvb, nt * 16, 64, k0, lane);
          kp[nt] = WMMA_BF16(Af, Bf, kp[nt]);
        }
      }
      #pragma unroll
      for (int i = 0; i < 8; ++i) {
        int b = b0 + mt * 16 + half * 8 + i;
        size_t qb = ((size_t)b * NHEAD + h) * HS;
        float dp = 0.f, dn = 0.f;
        #pragma unroll
        for (int nt = 0; nt < 4; ++nt) {
          float kval = kp[nt][i];
          dp += qp[qb + nt * 16 + n15] * kval;
          dn += qn[qb + nt * 16 + n15] * kval;
        }
        #pragma unroll
        for (int off = 1; off < 16; off <<= 1) {
          dp += __shfl_xor(dp, off, 32);
          dn += __shfl_xor(dn, off, 32);
        }
        if (n15 == 0) {
          size_t so = ((size_t)b * NHEAD + h) * RF + r;
          sp[so] = dp * scale;
          sn[so] = dn * scale;
        }
      }
    }
    __builtin_amdgcn_wave_barrier();
  }
}

// ---------- kernel 3: r-softmax + output ----------
__global__ __launch_bounds__(256) void koha_out(
    const float* __restrict__ sp, const float* __restrict__ sn,
    const u16* __restrict__ avt, float* __restrict__ out) {
  extern __shared__ char lds_raw[];
  const int tid  = threadIdx.x;
  const int lane = tid & 31;
  const int wave = tid >> 5;
  u16* att = (u16*)lds_raw + wave * 2 * (16 * 72);           // [2][16][72]

  const int task = blockIdx.x * 8 + wave;
  const int bt = task / NHEAD, h = task % NHEAD;
  const int b0 = bt * 16;
  const int n15 = lane & 15;
  const int half = (lane >> 4) & 1;

  // lanes 0-15: pos rows; lanes 16-31: neg rows
  const float* srow = (half ? sn : sp) + ((size_t)(b0 + n15) * NHEAD + h) * RF;
  float v[RF];
  #pragma unroll
  for (int g = 0; g < 16; ++g) {
    float4 t = *(const float4*)(srow + g * 4);
    v[g*4+0]=t.x; v[g*4+1]=t.y; v[g*4+2]=t.z; v[g*4+3]=t.w;
  }
  float mx = v[0];
  #pragma unroll
  for (int j = 1; j < RF; ++j) mx = fmaxf(mx, v[j]);
  float s = 0.f;
  #pragma unroll
  for (int j = 0; j < RF; ++j) { v[j] = __expf(v[j] - mx); s += v[j]; }
  float inv = 1.f / s;
  u16* my = att + half * (16 * 72);
  #pragma unroll
  for (int g = 0; g < 16; ++g) {
    ushort4 o;
    o.x = f2bf(v[g*4+0]*inv); o.y = f2bf(v[g*4+1]*inv);
    o.z = f2bf(v[g*4+2]*inv); o.w = f2bf(v[g*4+3]*inv);
    *(ushort4*)(my + n15 * 72 + g * 4) = o;
  }
  __syncthreads();

  const u16* avb = avt + (size_t)h * HS * RF;                // [n][r]
  #pragma unroll
  for (int p = 0; p < 2; ++p) {
    v8f acc[4];
    #pragma unroll
    for (int nt = 0; nt < 4; ++nt) acc[nt] = vzero8();
    u16* atile = att + p * (16 * 72);
    #pragma unroll
    for (int k0 = 0; k0 < RF; k0 += 32) {
      v16bf Af = load_frag(atile, 0, 72, k0, lane);
      #pragma unroll
      for (int nt = 0; nt < 4; ++nt) {
        v16bf Bf = load_frag(avb, nt * 16, RF, k0, lane);
        acc[nt] = WMMA_BF16(Af, Bf, acc[nt]);
      }
    }
    float* ob = out + (size_t)p * BTOT * EMB;
    #pragma unroll
    for (int nt = 0; nt < 4; ++nt)
      #pragma unroll
      for (int i = 0; i < 8; ++i)
        ob[(size_t)(b0 + half * 8 + i) * EMB + h * HS + nt * 16 + n15] = acc[nt][i];
  }
}

// ---------- launch ----------
extern "C" void kernel_launch(void* const* d_in, const int* in_sizes, int n_in,
                              void* d_out, int out_size, void* d_ws, size_t ws_size,
                              hipStream_t stream) {
  const float* x  = (const float*)d_in[0];
  const float* z  = (const float*)d_in[1];
  const float* qk = (const float*)d_in[2];
  const float* qv = (const float*)d_in[3];
  const float* kk = (const float*)d_in[4];
  const float* kv = (const float*)d_in[5];
  const float* av = (const float*)d_in[6];
  float* out = (float*)d_out;
  char* ws = (char*)d_ws;

  // workspace layout (bytes)
  u16* kk_bf = (u16*)(ws);                         // 134,217,728
  u16* qk_bf = (u16*)(ws + 134217728);             //   2,097,152
  u16* qvt   = (u16*)(ws + 136314880);             //     131,072
  u16* kvt   = (u16*)(ws + 136445952);             //   8,388,608
  u16* avt   = (u16*)(ws + 144834560);             //     131,072
  float* qpb = (float*)(ws + 144965632);           //   8,388,608
  float* qnb = (float*)(ws + 153354240);           //   8,388,608
  float* spb = (float*)(ws + 161742848);           //   8,388,608
  float* snb = (float*)(ws + 170131456);           //   8,388,608

  cvt_bf16_v4<<<8192, 256, 0, stream>>>((const float4*)kk, (ushort4*)kk_bf,
                                        (long)NHEAD * RF * HS * EMB / 4);
  cvt_bf16_v4<<<1024, 256, 0, stream>>>((const float4*)qk, (ushort4*)qk_bf,
                                        (long)NHEAD * HS * EMB / 4);
  transpose64_bf16<<<16,   256, 0, stream>>>(qv, qvt);
  transpose64_bf16<<<1024, 256, 0, stream>>>(kv, kvt);
  transpose64_bf16<<<16,   256, 0, stream>>>(av, avt);

  koha_query<<<BTOT / 16, 256, 16 * 1032 * 2 + 8 * 2 * 16 * 72 * 2, stream>>>(
      x, qk_bf, qvt, qpb, qnb);
  koha_scores<<<dim3(BTOT / 64, RF), 256, 64 * 1032 * 2 + 8 * 64 * 68 * 4, stream>>>(
      z, kk_bf, kvt, qpb, qnb, spb, snb);
  koha_out<<<(BTOT / 16) * NHEAD / 8, 256, 8 * 2 * 16 * 72 * 2, stream>>>(
      spb, snb, avt, out);
}
